// MultiHeadSelfAttention_12060268167485
// MI455X (gfx1250) — compile-verified, matrix operand order checked by reading, not run on hardware
//
#include <hip/hip_runtime.h>

// ---------------------------------------------------------------------------
// MI455X (gfx1250) multi-head self-attention, bf16 WMMA everywhere.
// B=4, S=2048, D=1024, H=16, Dh=64.
// GEMMs: double-buffered async Global->LDS staging (ASYNCcnt) when available.
// ---------------------------------------------------------------------------

typedef __attribute__((ext_vector_type(16))) __bf16        v16bf;
typedef __attribute__((ext_vector_type(8)))  float         v8f;
typedef __attribute__((ext_vector_type(4)))  unsigned int  v4u;
typedef __attribute__((ext_vector_type(4)))  int           v4i;

#define EMBED 1024
#define SEQ   2048
#define BATCH 4
#define HEADS 16
#define HDIM  64
#define NQKV  3072

#if __has_builtin(__builtin_amdgcn_global_load_async_to_lds_b128) && \
    __has_builtin(__builtin_amdgcn_s_wait_asynccnt)
#define USE_ASYNC_LDS 1
typedef __attribute__((address_space(1))) v4i av1_v4i;   // global (prints __device__)
typedef __attribute__((address_space(3))) v4i av3_v4i;   // LDS    (prints __shared__)
#endif

union FragU {
  v16bf          v;
  v4u            q[2];
  unsigned short u[16];
};

__device__ __forceinline__ unsigned short f32_to_bf16(float f) {
  unsigned int x = __float_as_uint(f);
  x += 0x7fffu + ((x >> 16) & 1u);     // round-to-nearest-even
  return (unsigned short)(x >> 16);
}

// Load one 16x32 bf16 A/B fragment for lane-group g (lane>=16).
// Element pattern (CDNA5 ISA 7.12.2): e<8 -> K = 8g+e ; e>=8 -> K = 16+8g+(e-8).
// Two 16-byte chunks at offsets 8g and 16+8g (in bf16 units).
__device__ __forceinline__ v16bf load_frag(const unsigned short* p, int g) {
  FragU f;
  f.q[0] = *(const v4u*)(p + 8 * g);
  f.q[1] = *(const v4u*)(p + 16 + 8 * g);
  return f.v;
}

__device__ __forceinline__ v8f wmma_bf16(v16bf a, v16bf b, v8f c) {
  return __builtin_amdgcn_wmma_f32_16x16x32_bf16(
      /*neg_a=*/false, a, /*neg_b=*/false, b,
      /*c_mod=*/(short)0, c, /*reuse_a=*/false, /*reuse_b=*/false);
}

// ---------------------------------------------------------------------------
// Elementwise fp32 -> bf16
// ---------------------------------------------------------------------------
__global__ void cvt_bf16_kernel(const float* __restrict__ src,
                                unsigned short* __restrict__ dst, int n) {
  int i = blockIdx.x * blockDim.x + threadIdx.x;
  if (i < n) dst[i] = f32_to_bf16(src[i]);
}

// dst[n*K + k] = bf16(src[k*N + n])  (weights -> N-major bf16 for B-fragments)
__global__ void transpose_bf16_kernel(const float* __restrict__ src,
                                      unsigned short* __restrict__ dst,
                                      int N, int K) {
  int n = blockIdx.x * blockDim.x + threadIdx.x;
  int k = blockIdx.y;
  if (n < N) dst[(size_t)n * K + k] = f32_to_bf16(src[(size_t)k * N + n]);
}

// ---------------------------------------------------------------------------
// Stage one 128x32 bf16 tile pair into LDS (one 64B row slice per thread).
// Async path: GLOBAL_LOAD_ASYNC_TO_LDS_B128, no VGPR round-trip, ASYNCcnt.
// ---------------------------------------------------------------------------
__device__ __forceinline__ void stage_tiles(unsigned short* As, unsigned short* Bs,
                                            const unsigned short* aRow,
                                            const unsigned short* bRow,
                                            int t, int kt) {
#ifdef USE_ASYNC_LDS
#pragma unroll
  for (int u = 0; u < 4; ++u) {
    v4i* ga = (v4i*)(aRow + kt + u * 8);
    v4i* la = (v4i*)&As[t * 40 + u * 8];
    __builtin_amdgcn_global_load_async_to_lds_b128((av1_v4i*)ga, (av3_v4i*)la, 0, 0);
    v4i* gb = (v4i*)(bRow + kt + u * 8);
    v4i* lb = (v4i*)&Bs[t * 40 + u * 8];
    __builtin_amdgcn_global_load_async_to_lds_b128((av1_v4i*)gb, (av3_v4i*)lb, 0, 0);
  }
#else
#pragma unroll
  for (int u = 0; u < 4; ++u) {
    *(v4u*)&As[t * 40 + u * 8] = *(const v4u*)(aRow + kt + u * 8);
    *(v4u*)&Bs[t * 40 + u * 8] = *(const v4u*)(bRow + kt + u * 8);
  }
#endif
}

// ---------------------------------------------------------------------------
// Tiled WMMA GEMM:  C[M,N] = A[M,K](bf16) * B^T (Bm stored [N,K] bf16) + bias
// Block tile 128x128, 4 waves (128 thr), each wave 64x64 = 4x4 WMMA accums.
// Double-buffered LDS: copy tile k+1 overlaps the 16 WMMAs on tile k.
// MODE 0: QKV epilogue -> scatter to Q (scaled 1/sqrt(Dh)), K, V^T  (bf16)
// MODE 1: plain f32 output + bias
// ---------------------------------------------------------------------------
template <int MODE>
__global__ void __launch_bounds__(128)
gemm_wmma_kernel(const unsigned short* __restrict__ A,
                 const unsigned short* __restrict__ Bm,
                 const float* __restrict__ bias,
                 float* __restrict__ outF,
                 unsigned short* __restrict__ Qb,
                 unsigned short* __restrict__ Kb,
                 unsigned short* __restrict__ Vt,
                 int N, int K) {
  (void)outF; (void)Qb; (void)Kb; (void)Vt;

  // 2 x (128 rows x 32 bf16 padded to 40): 80B rows, 16B aligned slots.
  __shared__ v4u AsV[2][128 * 5];
  __shared__ v4u BsV[2][128 * 5];

  const int t    = threadIdx.x;
  const int w    = t >> 5;
  const int lane = t & 31;
  const int g    = lane >> 4;
  const int c16  = lane & 15;
  const int wm   = (w >> 1) * 64;
  const int wn   = (w & 1) * 64;
  const int bm   = blockIdx.y * 128;
  const int bn   = blockIdx.x * 128;

  v8f c[4][4];
#pragma unroll
  for (int i = 0; i < 4; ++i)
#pragma unroll
    for (int j = 0; j < 4; ++j)
#pragma unroll
      for (int r = 0; r < 8; ++r) c[i][j][r] = 0.0f;

  const unsigned short* aRow = A  + (size_t)(bm + t) * K;
  const unsigned short* bRow = Bm + (size_t)(bn + t) * K;

  // prologue: stage tile 0 into buffer 0
  stage_tiles((unsigned short*)AsV[0], (unsigned short*)BsV[0], aRow, bRow, t, 0);

  for (int kt = 0; kt < K; kt += 32) {
    const int buf = (kt >> 5) & 1;
#ifdef USE_ASYNC_LDS
    __builtin_amdgcn_s_wait_asynccnt(0);   // my tile-kt copies have landed
#endif
    __syncthreads();                       // everyone's copies landed; buf^1 free

    if (kt + 32 < K) {                     // overlap next-tile copy with WMMAs
      stage_tiles((unsigned short*)AsV[buf ^ 1], (unsigned short*)BsV[buf ^ 1],
                  aRow, bRow, t, kt + 32);
      __builtin_prefetch(aRow + kt + 64, 0, 1);
      __builtin_prefetch(bRow + kt + 64, 0, 1);
    }

    unsigned short* As = (unsigned short*)AsV[buf];
    unsigned short* Bs = (unsigned short*)BsV[buf];

    v16bf af[4], bfm[4];
#pragma unroll
    for (int mi = 0; mi < 4; ++mi)
      af[mi] = load_frag(&As[(wm + mi * 16 + c16) * 40], g);
#pragma unroll
    for (int ni = 0; ni < 4; ++ni)
      bfm[ni] = load_frag(&Bs[(wn + ni * 16 + c16) * 40], g);
#pragma unroll
    for (int mi = 0; mi < 4; ++mi)
#pragma unroll
      for (int ni = 0; ni < 4; ++ni)
        c[mi][ni] = wmma_bf16(af[mi], bfm[ni], c[mi][ni]);
  }

  float bv[4];
#pragma unroll
  for (int ni = 0; ni < 4; ++ni) bv[ni] = bias[bn + wn + ni * 16 + c16];

#pragma unroll
  for (int mi = 0; mi < 4; ++mi) {
#pragma unroll
    for (int ni = 0; ni < 4; ++ni) {
      const int gn = bn + wn + ni * 16 + c16;
#pragma unroll
      for (int r = 0; r < 8; ++r) {
        const int gm = bm + wm + mi * 16 + r + 8 * g;   // C layout: row=r+8*half
        const float val = c[mi][ni][r] + bv[ni];
        if constexpr (MODE == 1) {
          outF[(size_t)gm * N + gn] = val;
        } else {
          const int which = gn >> 10;       // 0=Q 1=K 2=V
          const int n1 = gn & 1023;
          const int h = n1 >> 6, d = n1 & 63;
          const int b = gm >> 11, s = gm & 2047;
          const size_t qk = (((size_t)(b * HEADS + h) * SEQ + s) << 6) + d;
          if (which == 0)      Qb[qk] = f32_to_bf16(val * 0.125f); // 1/sqrt(64)
          else if (which == 1) Kb[qk] = f32_to_bf16(val);
          else Vt[((size_t)(b * HEADS + h) * HDIM + d) * SEQ + s] = f32_to_bf16(val);
        }
      }
    }
  }
}

// ---------------------------------------------------------------------------
// Flash attention, transposed formulation (no LDS, no transposes):
//   S^T = K * Q^T   (A = K rows, B = Q cols)       -> stats per query at lane%16
//   O^T = V^T * P^T (A = V^T rows, B = P^T repack) -> same per-lane stats
// Block = 4 waves; each wave owns 16 queries, streams 32 keys / iteration.
// ---------------------------------------------------------------------------
__global__ void __launch_bounds__(128)
flash_attn_kernel(const unsigned short* __restrict__ Qb,
                  const unsigned short* __restrict__ Kb,
                  const unsigned short* __restrict__ Vt,
                  unsigned short* __restrict__ Ob) {
  const int bh   = blockIdx.x;
  const int w    = threadIdx.x >> 5;
  const int lane = threadIdx.x & 31;
  const int g    = lane >> 4;
  const int c16  = lane & 15;

  const unsigned short* Qh = Qb + (size_t)bh * SEQ * HDIM;
  const unsigned short* Kh = Kb + (size_t)bh * SEQ * HDIM;
  const unsigned short* Vh = Vt + (size_t)bh * HDIM * SEQ;

  const int q0 = blockIdx.y * 64 + w * 16;

  // Q fragments (B operand), reused over every key iteration
  v16bf qf[2];
  {
    const unsigned short* qrow = Qh + (size_t)(q0 + c16) * HDIM;
    qf[0] = load_frag(qrow, g);        // d in [0,32)
    qf[1] = load_frag(qrow + 32, g);   // d in [32,64)
  }

  float m = -3.0e38f, l = 0.0f;
  v8f o[4];
#pragma unroll
  for (int dt = 0; dt < 4; ++dt)
#pragma unroll
    for (int r = 0; r < 8; ++r) o[dt][r] = 0.0f;

  for (int k0 = 0; k0 < SEQ; k0 += 32) {
    // K fragments (A operand): 2 key subtiles x 2 d-steps
    v16bf ka[2][2];
#pragma unroll
    for (int ks = 0; ks < 2; ++ks) {
      const unsigned short* krow = Kh + (size_t)(k0 + ks * 16 + c16) * HDIM;
      ka[ks][0] = load_frag(krow, g);
      ka[ks][1] = load_frag(krow + 32, g);
    }

    v8f st0, st1;
#pragma unroll
    for (int r = 0; r < 8; ++r) { st0[r] = 0.0f; st1[r] = 0.0f; }
    st0 = wmma_bf16(ka[0][0], qf[0], st0);   // keys k0..k0+15
    st0 = wmma_bf16(ka[0][1], qf[1], st0);
    st1 = wmma_bf16(ka[1][0], qf[0], st1);   // keys k0+16..k0+31
    st1 = wmma_bf16(ka[1][1], qf[1], st1);

    // online softmax; per-query stats live at lane%16 (pair lane^16)
    float vmax = st0[0];
#pragma unroll
    for (int r = 0; r < 8; ++r) {
      vmax = fmaxf(vmax, st0[r]);
      vmax = fmaxf(vmax, st1[r]);
    }
    vmax = fmaxf(vmax, __shfl_xor(vmax, 16, 32));
    const float mnew  = fmaxf(m, vmax);
    const float alpha = __expf(m - mnew);

    float p0[8], p1[8], rs = 0.0f;
#pragma unroll
    for (int r = 0; r < 8; ++r) {
      p0[r] = __expf(st0[r] - mnew);
      p1[r] = __expf(st1[r] - mnew);
      rs += p0[r] + p1[r];
    }
    rs += __shfl_xor(rs, 16, 32);
    l = l * alpha + rs;
    m = mnew;

#pragma unroll
    for (int dt = 0; dt < 4; ++dt)
#pragma unroll
      for (int r = 0; r < 8; ++r) o[dt][r] *= alpha;

    // Register repack: score C-layout -> B-fragment (k-contraction) layout.
    FragU pb;
#pragma unroll
    for (int e = 0; e < 8; ++e) {
      pb.u[e]     = f32_to_bf16(p0[e]);   // kc = 8g+e
      pb.u[e + 8] = f32_to_bf16(p1[e]);   // kc = 16+8g+e
    }

    // O^T accumulate: A = V^T rows (d), contraction over 32 keys
#pragma unroll
    for (int dt = 0; dt < 4; ++dt) {
      const unsigned short* vrow = Vh + (size_t)(dt * 16 + c16) * SEQ + k0;
      v16bf va = load_frag(vrow, g);
      o[dt] = wmma_bf16(va, pb.v, o[dt]);
    }
  }

  const float invl = (l > 0.0f) ? (1.0f / l) : 0.0f;
  const int b = bh >> 4, h = bh & 15;
  const int qg = q0 + c16;
#pragma unroll
  for (int dt = 0; dt < 4; ++dt)
#pragma unroll
    for (int r = 0; r < 8; ++r) {
      const int d = dt * 16 + r + 8 * g;
      Ob[(size_t)(b * SEQ + qg) * EMBED + h * HDIM + d] =
          f32_to_bf16(o[dt][r] * invl);
    }
}

// ---------------------------------------------------------------------------
// Launch
// ---------------------------------------------------------------------------
extern "C" void kernel_launch(void* const* d_in, const int* in_sizes, int n_in,
                              void* d_out, int out_size, void* d_ws,
                              size_t ws_size, hipStream_t stream) {
  (void)in_sizes; (void)n_in; (void)out_size; (void)ws_size;

  const float* x    = (const float*)d_in[0];
  const float* Wqkv = (const float*)d_in[1];
  const float* bqkv = (const float*)d_in[2];
  const float* Wout = (const float*)d_in[3];
  const float* bout = (const float*)d_in[4];
  float* out = (float*)d_out;

  unsigned short* ws  = (unsigned short*)d_ws;
  unsigned short* Xb  = ws;                       // [8192,1024] bf16
  unsigned short* Wqt = Xb  + (size_t)8192 * 1024;  // [3072,1024] bf16 (W_qkv^T)
  unsigned short* Wot = Wqt + (size_t)3072 * 1024;  // [1024,1024] bf16 (W_out^T)
  unsigned short* Qb  = Wot + (size_t)1024 * 1024;  // [B,H,S,Dh] bf16 (pre-scaled)
  unsigned short* Kb  = Qb  + (size_t)64 * SEQ * HDIM;
  unsigned short* Vt  = Kb  + (size_t)64 * SEQ * HDIM;  // [B,H,Dh,S]
  unsigned short* Ob  = Vt  + (size_t)64 * SEQ * HDIM;  // [8192,1024] bf16

  const int nX = BATCH * SEQ * EMBED;  // 8388608
  cvt_bf16_kernel<<<(nX + 255) / 256, 256, 0, stream>>>(x, Xb, nX);
  transpose_bf16_kernel<<<dim3(NQKV / 128, EMBED), 128, 0, stream>>>(
      Wqkv, Wqt, NQKV, EMBED);
  transpose_bf16_kernel<<<dim3(EMBED / 128, EMBED), 128, 0, stream>>>(
      Wout, Wot, EMBED, EMBED);

  // QKV projection: M=8192, N=3072, K=1024
  gemm_wmma_kernel<0><<<dim3(NQKV / 128, (BATCH * SEQ) / 128), 128, 0, stream>>>(
      Xb, Wqt, bqkv, nullptr, Qb, Kb, Vt, NQKV, EMBED);

  // Flash attention: grid (B*H, S/64)
  flash_attn_kernel<<<dim3(BATCH * HEADS, SEQ / 64), 128, 0, stream>>>(
      Qb, Kb, Vt, Ob);

  // Output projection: M=8192, N=1024, K=1024
  gemm_wmma_kernel<1><<<dim3(EMBED / 128, (BATCH * SEQ) / 128), 128, 0, stream>>>(
      Ob, Wot, bout, out, nullptr, nullptr, nullptr, EMBED, EMBED);
}
